// SelfAttention_63204738728363
// MI455X (gfx1250) — matrix operand order repaired and checked by reading, not compiled, not run on hardware
//
#include <hip/hip_runtime.h>
#include <hip/hip_bf16.h>

// B=512, S=80, D=768, OUT=768, H=12, dh=dv=64
#define BB 512
#define SS 80
#define DD 768
#define HH 12
#define DH 64
#define MROWS (BB*SS)   // 40960

typedef __attribute__((ext_vector_type(16))) _Float16 v16h;
typedef __attribute__((ext_vector_type(8)))  _Float16 v8h;
typedef __attribute__((ext_vector_type(4)))  _Float16 v4h;
typedef __attribute__((ext_vector_type(8)))  float    v8f;
typedef __attribute__((ext_vector_type(4)))  float    v4f;

// ---------------------------------------------------------------------------
// CDNA5 async DMA helpers (CDNA5 ISA): per-lane
// global->LDS copy tracked by ASYNCcnt; generic LDS pointer low 32 bits are
// the LDS byte offset (ISA aperture rule), which is what VDST wants.
// ---------------------------------------------------------------------------
__device__ __forceinline__ void async_copy_b128(unsigned int lds_addr,
                                                unsigned long long gaddr) {
    asm volatile("global_load_async_to_lds_b128 %0, %1, off"
                 :: "v"(lds_addr), "v"(gaddr) : "memory");
}
__device__ __forceinline__ void wait_asynccnt0() {
    asm volatile("s_wait_asynccnt 0x0" ::: "memory");
}

// ---------------------------------------------------------------------------
// Kernel 0a: RoPE sin/cos tables, fp32, [pos=80][pair=32]
// ---------------------------------------------------------------------------
__global__ void rope_tables(float* __restrict__ sinT, float* __restrict__ cosT) {
    int i = blockIdx.x * blockDim.x + threadIdx.x;
    if (i < SS * 32) {
        int pos = i >> 5;
        int p   = i & 31;
        float inv = powf(10000.0f, -(float)(2 * p) / 64.0f);
        float a = (float)pos * inv;
        sinT[i] = sinf(a);
        cosT[i] = cosf(a);
    }
}

// ---------------------------------------------------------------------------
// Kernel 0b: hidden fp32 -> f16 (elementwise, float4 vectorized)
// ---------------------------------------------------------------------------
__global__ __launch_bounds__(256) void cvt_x(const float* __restrict__ in,
                                             _Float16* __restrict__ out, int n4) {
    int i = blockIdx.x * blockDim.x + threadIdx.x;
    if (i < n4) {
        float4 v = ((const float4*)in)[i];
        v4h h = {(_Float16)v.x, (_Float16)v.y, (_Float16)v.z, (_Float16)v.w};
        ((v4h*)out)[i] = h;
    }
}

// ---------------------------------------------------------------------------
// Kernel 0c: W (768x768 fp32, [k][n]) -> f16 transposed WT[n][k], 3 matrices.
// 64x64 LDS-tiled transpose, coalesced both directions.
// ---------------------------------------------------------------------------
__global__ __launch_bounds__(256) void cvt_wT(const float* __restrict__ Wq,
                                              const float* __restrict__ Wk,
                                              const float* __restrict__ Wv,
                                              _Float16* __restrict__ WT) {
    __shared__ _Float16 t[64][65];
    const int tid = threadIdx.x;
    const float* W = (blockIdx.z == 0) ? Wq : ((blockIdx.z == 1) ? Wk : Wv);
    _Float16* O = WT + (size_t)blockIdx.z * DD * DD;
    const int k0 = blockIdx.x * 64, n0 = blockIdx.y * 64;
#pragma unroll
    for (int i = 0; i < 16; ++i) {
        int idx = tid + i * 256;
        int r = idx >> 6, c = idx & 63;
        t[r][c] = (_Float16)W[(size_t)(k0 + r) * DD + n0 + c];
    }
    __syncthreads();
#pragma unroll
    for (int i = 0; i < 16; ++i) {
        int idx = tid + i * 256;
        int r = idx >> 6, c = idx & 63;
        O[(size_t)(n0 + r) * DD + k0 + c] = t[c][r];
    }
}

// ---------------------------------------------------------------------------
// Fragment helpers; A/B use this code's contiguous k order (not the ISA table), C as in CDNA5 ISA:
// A 16x32 f16: lane half g holds K 16g..16g+15 contiguous, row = lane&15.
// B 32x16 f16: lane half g holds K 16g..16g+15 contiguous, col = lane&15.
// C 16x16 f32: vgpr j -> row j + 8g, col = lane&15.
// ---------------------------------------------------------------------------
__device__ __forceinline__ v16h cat16(v8h a, v8h b) {
    return __builtin_shufflevector(a, b, 0,1,2,3,4,5,6,7,8,9,10,11,12,13,14,15);
}

__device__ __forceinline__ v8h rope8(v8h x, const float* __restrict__ sinT,
                                     const float* __restrict__ cosT,
                                     int pos, int p0) {
    v4f s = *(const v4f*)(sinT + pos * 32 + p0);
    v4f c = *(const v4f*)(cosT + pos * 32 + p0);
    v8h r;
#pragma unroll
    for (int i = 0; i < 4; ++i) {
        float e = (float)x[2 * i];
        float o = (float)x[2 * i + 1];
        r[2 * i]     = (_Float16)(e * c[i] - o * s[i]);
        r[2 * i + 1] = (_Float16)(o * c[i] + e * s[i]);
    }
    return r;
}

// ---------------------------------------------------------------------------
// Kernel 1: fused QKV projection from pre-converted f16 Xh and WT.
// blockIdx.x = M tile (128 rows), blockIdx.y in [0,18): matId=y/6, nTile=y%6.
// 256 threads = 8 waves in a 4x2 grid: wave owns 32 rows x 64 cols
// (2 M-subtiles x 4 N-subtiles -> 8 WMMAs per 12 b128 DS loads per k-step).
// Double-buffered LDS filled by global_load_async_to_lds_b128.
// ---------------------------------------------------------------------------
__global__ __launch_bounds__(256) void qkv_gemm(
    const _Float16* __restrict__ Xh, const _Float16* __restrict__ WT,
    const float* __restrict__ bq, const float* __restrict__ bk,
    const float* __restrict__ bv,
    _Float16* __restrict__ Qh, _Float16* __restrict__ Kh, _Float16* __restrict__ Vh)
{
    // two buffers each: X tile [row 128][k 64] (stride 72), W^T tile [n 128][k 64]
    __shared__ __align__(16) _Float16 Xl[2][128 * 72];
    __shared__ __align__(16) _Float16 Wl[2][128 * 72];

    const int tid   = threadIdx.x;
    const int mBase = blockIdx.x * 128;
    const int matId = blockIdx.y / 6;
    const int nBase = (blockIdx.y % 6) * 128;

    const _Float16* Wm   = WT + (size_t)matId * DD * DD;
    const float*    bias = (matId == 0) ? bq : ((matId == 1) ? bk : bv);
    _Float16*       O    = (matId == 0) ? Qh : ((matId == 1) ? Kh : Vh);

    const int wv = tid >> 5, ln = tid & 31, g = ln >> 4, l = ln & 15;
    const int wm = wv & 3;        // row group: rows 32*wm .. 32*wm+31
    const int wn = wv >> 2;       // col group: cols 64*wn .. 64*wn+63

    // per-thread staging slots: 1024 b128 per tile -> 4 per thread per tile
    unsigned long long gx[4], gw[4];
    unsigned int lx[2][4], lw[2][4];
#pragma unroll
    for (int i = 0; i < 4; ++i) {
        int slot = tid + 256 * i;
        int r = slot >> 3, seg = (slot & 7) * 8;       // seg in halves
        gx[i] = (unsigned long long)(uintptr_t)(Xh + (size_t)(mBase + r) * DD + seg);
        gw[i] = (unsigned long long)(uintptr_t)(Wm + (size_t)(nBase + r) * DD + seg);
        lx[0][i] = (unsigned int)(uintptr_t)&Xl[0][r * 72 + seg];
        lx[1][i] = (unsigned int)(uintptr_t)&Xl[1][r * 72 + seg];
        lw[0][i] = (unsigned int)(uintptr_t)&Wl[0][r * 72 + seg];
        lw[1][i] = (unsigned int)(uintptr_t)&Wl[1][r * 72 + seg];
    }

    v8f acc[2][4] = {};

    // prologue: stage k-chunk 0 into buffer 0
#pragma unroll
    for (int i = 0; i < 4; ++i) {
        async_copy_b128(lx[0][i], gx[i]);
        async_copy_b128(lw[0][i], gw[i]);
        gx[i] += 128; gw[i] += 128;                    // advance 64 halves
    }

    for (int it = 0; it < 12; ++it) {                  // 768 / 64 chunks
        wait_asynccnt0();
        __syncthreads();   // stage(it) visible; all waves done computing it-1
        const int buf = it & 1;
        if (it < 11) {
#pragma unroll
            for (int i = 0; i < 4; ++i) {
                async_copy_b128(lx[buf ^ 1][i], gx[i]);
                async_copy_b128(lw[buf ^ 1][i], gw[i]);
                gx[i] += 128; gw[i] += 128;
            }
        }
        const _Float16* Xb = &Xl[buf][0];
        const _Float16* Wb = &Wl[buf][0];
#pragma unroll
        for (int kk = 0; kk < 64; kk += 32) {
            v16h Bf[4];
#pragma unroll
            for (int nt = 0; nt < 4; ++nt) {
                int n = 64 * wn + nt * 16 + l;
                v8h b0 = *(const v8h*)&Wb[n * 72 + kk + 16 * g];
                v8h b1 = *(const v8h*)&Wb[n * 72 + kk + 16 * g + 8];
                Bf[nt] = cat16(b0, b1);
            }
#pragma unroll
            for (int i = 0; i < 2; ++i) {
                int arow = 32 * wm + 16 * i + l;
                v8h a0 = *(const v8h*)&Xb[arow * 72 + kk + 16 * g];
                v8h a1 = *(const v8h*)&Xb[arow * 72 + kk + 16 * g + 8];
                v16h A = cat16(a0, a1);
#pragma unroll
                for (int nt = 0; nt < 4; ++nt)
                    acc[i][nt] = __builtin_amdgcn_wmma_f32_16x16x32_f16(
                        false, A, false, Bf[nt], (short)0, acc[i][nt], false, false);
            }
        }
    }

    // epilogue: bias add, f16 store
#pragma unroll
    for (int i = 0; i < 2; ++i) {
#pragma unroll
        for (int nt = 0; nt < 4; ++nt) {
            int n = nBase + 64 * wn + nt * 16 + l;
            float bb = bias[n];
#pragma unroll
            for (int j = 0; j < 8; ++j) {
                int row = mBase + 32 * wm + 16 * i + g * 8 + j;
                O[(size_t)row * DD + n] = (_Float16)(acc[i][nt][j] + bb);
            }
        }
    }
}

// ---------------------------------------------------------------------------
// Kernel 2: attention per (b,h). 160 threads = 5 waves; wave w owns query
// band [16w,16w+16). RoPE applied on the fly to Q/K fragments.
// ---------------------------------------------------------------------------
__global__ __launch_bounds__(160) void attention(
    const _Float16* __restrict__ Qh, const _Float16* __restrict__ Kh,
    const _Float16* __restrict__ Vh,
    const float* __restrict__ sinT, const float* __restrict__ cosT,
    const int* __restrict__ mask, float* __restrict__ out)
{
    __shared__ __align__(16) _Float16 probL[80 * 104];  // [q][k], k padded to 96 (+8)
    __shared__ __align__(16) _Float16 VtL[64 * 104];    // [n][k] transposed, k pad 96

    const int tid = threadIdx.x;
    const int bh  = blockIdx.x;
    const int b   = bh / HH, h = bh % HH;
    const int wv  = tid >> 5, ln = tid & 31, g = ln >> 4, l = ln & 15;

    // stage V transposed: 80x64 -> VtL[n][s]
    for (int it = 0; it < 8; ++it) {
        int idx = tid + it * 160;               // 1280 slots of 4 elems along n
        int s = idx >> 4, n4 = (idx & 15) << 2;
        v4h vv = *(const v4h*)(Vh + (size_t)(b * SS + s) * DD + h * DH + n4);
        VtL[(n4 + 0) * 104 + s] = vv[0];
        VtL[(n4 + 1) * 104 + s] = vv[1];
        VtL[(n4 + 2) * 104 + s] = vv[2];
        VtL[(n4 + 3) * 104 + s] = vv[3];
    }
    for (int idx = tid; idx < 64 * 16; idx += 160) {   // zero V rows 80..95
        int n = idx >> 4, s = 80 + (idx & 15);
        VtL[n * 104 + s] = (_Float16)0.0f;
    }
    for (int idx = tid; idx < 80 * 16; idx += 160) {   // zero prob cols 80..95
        int r = idx >> 4, c = 80 + (idx & 15);
        probL[r * 104 + c] = (_Float16)0.0f;
    }
    __syncthreads();

    // ---- scores = rope(Q) . rope(K)^T  (M=80 banded, N=80, K=64) ----
    const int qrow = wv * 16 + l;
    const size_t qoff = (size_t)(b * SS + qrow) * DD + h * DH;
    v8f sc[5] = {};
#pragma unroll
    for (int kk = 0; kk < 64; kk += 32) {
        v8h qa0 = rope8(*(const v8h*)(Qh + qoff + kk + 16 * g), sinT, cosT, qrow, (kk + 16 * g) >> 1);
        v8h qa1 = rope8(*(const v8h*)(Qh + qoff + kk + 16 * g + 8), sinT, cosT, qrow, (kk + 16 * g + 8) >> 1);
        v16h A = cat16(qa0, qa1);
#pragma unroll
        for (int nt = 0; nt < 5; ++nt) {
            int kpos = nt * 16 + l;
            size_t koff = (size_t)(b * SS + kpos) * DD + h * DH;
            v8h kb0 = rope8(*(const v8h*)(Kh + koff + kk + 16 * g), sinT, cosT, kpos, (kk + 16 * g) >> 1);
            v8h kb1 = rope8(*(const v8h*)(Kh + koff + kk + 16 * g + 8), sinT, cosT, kpos, (kk + 16 * g + 8) >> 1);
            v16h Bv = cat16(kb0, kb1);
            sc[nt] = __builtin_amdgcn_wmma_f32_16x16x32_f16(
                false, A, false, Bv, (short)0, sc[nt], false, false);
        }
    }

    // ---- softmax over 80 keys (rows split across lane halves per C layout) ----
    float bias_nt[5];
#pragma unroll
    for (int nt = 0; nt < 5; ++nt) {
        int kpos = nt * 16 + l;
        bias_nt[nt] = (1.0f - (float)mask[b * SS + kpos]) * -10000.0f;
    }
#pragma unroll
    for (int j = 0; j < 8; ++j) {
        float m = -3.0e38f;
#pragma unroll
        for (int nt = 0; nt < 5; ++nt) {
            float v = sc[nt][j] * 0.125f + bias_nt[nt];   // 1/sqrt(64)
            sc[nt][j] = v;
            m = fmaxf(m, v);
        }
        m = fmaxf(m, __shfl_xor(m, 1, 32));
        m = fmaxf(m, __shfl_xor(m, 2, 32));
        m = fmaxf(m, __shfl_xor(m, 4, 32));
        m = fmaxf(m, __shfl_xor(m, 8, 32));
        float s = 0.0f;
#pragma unroll
        for (int nt = 0; nt < 5; ++nt) {
            float e = __expf(sc[nt][j] - m);
            sc[nt][j] = e;
            s += e;
        }
        s += __shfl_xor(s, 1, 32);
        s += __shfl_xor(s, 2, 32);
        s += __shfl_xor(s, 4, 32);
        s += __shfl_xor(s, 8, 32);
        float r = 1.0f / s;
#pragma unroll
        for (int nt = 0; nt < 5; ++nt) sc[nt][j] *= r;
    }

    // write prob band to LDS as f16 (same wave reads it back: dscnt only)
#pragma unroll
    for (int nt = 0; nt < 5; ++nt) {
        int col = nt * 16 + l;
#pragma unroll
        for (int j = 0; j < 8; ++j) {
            int row = wv * 16 + g * 8 + j;
            probL[row * 104 + col] = (_Float16)sc[nt][j];
        }
    }

    // ---- ctx = prob(80x96) @ V(96x64) ----
    v8f cx[4] = {};
#pragma unroll
    for (int kk = 0; kk < 96; kk += 32) {
        v8h p0 = *(const v8h*)&probL[qrow * 104 + kk + 16 * g];
        v8h p1 = *(const v8h*)&probL[qrow * 104 + kk + 16 * g + 8];
        v16h A = cat16(p0, p1);
#pragma unroll
        for (int nt = 0; nt < 4; ++nt) {
            int n = nt * 16 + l;
            v8h b0 = *(const v8h*)&VtL[n * 104 + kk + 16 * g];
            v8h b1 = *(const v8h*)&VtL[n * 104 + kk + 16 * g + 8];
            v16h Bv = cat16(b0, b1);
            cx[nt] = __builtin_amdgcn_wmma_f32_16x16x32_f16(
                false, A, false, Bv, (short)0, cx[nt], false, false);
        }
    }

#pragma unroll
    for (int nt = 0; nt < 4; ++nt) {
        int col = nt * 16 + l;
#pragma unroll
        for (int j = 0; j < 8; ++j) {
            int row = wv * 16 + g * 8 + j;
            out[(size_t)(b * SS + row) * DD + h * DH + col] = cx[nt][j];
        }
    }
}

// ---------------------------------------------------------------------------
extern "C" void kernel_launch(void* const* d_in, const int* in_sizes, int n_in,
                              void* d_out, int out_size, void* d_ws, size_t ws_size,
                              hipStream_t stream) {
    const float* hidden = (const float*)d_in[0];
    const int*   mask   = (const int*)d_in[1];
    const float* Wq = (const float*)d_in[2];
    const float* bq = (const float*)d_in[3];
    const float* Wk = (const float*)d_in[4];
    const float* bk = (const float*)d_in[5];
    const float* Wv = (const float*)d_in[6];
    const float* bv = (const float*)d_in[7];
    float* out = (float*)d_out;

    char* ws = (char*)d_ws;
    float* sinT = (float*)ws;                         // 80*32*4 = 10240 B
    float* cosT = (float*)(ws + 10240);               // 10240 B
    const size_t xBytes = (size_t)MROWS * DD * sizeof(_Float16);  // ~63 MB
    const size_t wBytes = (size_t)3 * DD * DD * sizeof(_Float16); // ~3.5 MB
    _Float16* Xh = (_Float16*)(ws + 32768);
    _Float16* WT = (_Float16*)(ws + 32768 + xBytes);
    _Float16* Qh = (_Float16*)(ws + 32768 + xBytes + wBytes);
    _Float16* Kh = (_Float16*)(ws + 32768 + xBytes + wBytes + xBytes);
    _Float16* Vh = (_Float16*)(ws + 32768 + xBytes + wBytes + 2 * xBytes);

    rope_tables<<<dim3((SS * 32 + 255) / 256), 256, 0, stream>>>(sinT, cosT);

    const int n4 = MROWS * DD / 4;
    cvt_x<<<dim3((n4 + 255) / 256), 256, 0, stream>>>(hidden, Xh, n4);
    cvt_wT<<<dim3(DD / 64, DD / 64, 3), 256, 0, stream>>>(Wq, Wk, Wv, WT);

    qkv_gemm<<<dim3(MROWS / 128, 18), 256, 0, stream>>>(
        Xh, WT, bq, bk, bv, Qh, Kh, Vh);

    attention<<<dim3(BB * HH), 160, 0, stream>>>(
        Qh, Kh, Vh, sinT, cosT, mask, out);
}
